// RTDETRHead_80290118632056
// MI455X (gfx1250) — matrix operand order repaired and checked by reading, not compiled, not run on hardware
//
#include <hip/hip_runtime.h>

// ---------------------------------------------------------------------------
// RT-DETR encoder head for MI455X (gfx1250), wave32 + WMMA bf16.
// ---------------------------------------------------------------------------

typedef __attribute__((ext_vector_type(16))) __bf16 v16bf;
typedef __attribute__((ext_vector_type(8)))  float  v8f;

#define C_DIM       256
#define N_TOK       21504          // 128*128 + 64*64 + 32*32
#define NLVL3       16384
#define NLVL4       4096
#define NLVL5       1024
#define BATCH       16
#define NCLS        80
#define KQ          300
#define T_TILE      64             // tokens per block in fused kernel
#define ROWSTRIDE   264            // bf16 elements per LDS row (256 + 8 pad)

// ---- workspace layout (bytes) ----
#define WS_W1       0u             // 16 ct * 8 kt * 512 bf16 = 131072 B
#define WS_W2       131072u
#define WS_WSC      262144u        // 5 ct  -> 40960 B
#define WS_W3       303104u        // 1 ct  -> 8192 B
#define WS_WPROJ    311296u        // 131072 B
#define WS_SCORES   442368u        // B*N f32 = 1376256 B
#define WS_IDX      1818624u       // B*K int32 = 19200 B

// ---- d_out layout (float elements) ----
#define OFF_TGT     0u             // [16,300,256]
#define OFF_REF     1228800u       // [16,300,4]
#define OFF_LOG     1248000u       // [16,21504,80]
#define OFF_SBX     28773120u      // [16,21504,4]
#define OFF_IDX     30149376u      // [16,300]
#define OFF_SCR     30154176u      // [16,300]

__device__ __forceinline__ unsigned short f32_to_bf16(float f) {
    unsigned u = __float_as_uint(f);
    unsigned r = (u + 0x7FFFu + ((u >> 16) & 1u)) >> 16;
    return (unsigned short)r;
}
__device__ __forceinline__ unsigned fkey(float f) {
    unsigned u = __float_as_uint(f);
    return (u & 0x80000000u) ? ~u : (u | 0x80000000u);
}
__device__ __forceinline__ float unkey(unsigned k) {
    return __uint_as_float((k & 0x80000000u) ? (k ^ 0x80000000u) : ~k);
}
__device__ __forceinline__ float sigmoidf(float x) {
    return 1.0f / (1.0f + __expf(-x));
}

// A fragment (16x32 bf16) from LDS row-major tile; per ISA 7.12.2:
// lanes 0-15 (M=lane)  hold K = {0..7, 16..23}
// lanes 16-31 (M=l-16) hold K = {8..15, 24..31}
__device__ __forceinline__ v16bf load_a_frag(const unsigned short* lds,
                                             int rt, int kt, int lane) {
    int m    = lane & 15;
    int half = lane >> 4;
    const unsigned short* base = lds + (rt * 16 + m) * ROWSTRIDE + kt * 32 + half * 8;
    union { v16bf v; uint4 q[2]; } u;
    u.q[0] = *(const uint4*)(base);
    u.q[1] = *(const uint4*)(base + 16);
    return u.v;
}

// Load all 8 A fragments for one row tile (held in registers across ct sweep).
__device__ __forceinline__ void load_a_all(const unsigned short* lds,
                                           int rt, int lane, v16bf* a) {
#pragma unroll
    for (int kt = 0; kt < 8; ++kt) a[kt] = load_a_frag(lds, rt, kt, lane);
}

// B fragment (32x16 bf16), pre-repacked: contiguous 16 bf16 per lane.
__device__ __forceinline__ v16bf load_b_frag(const unsigned short* wf,
                                             int ct, int kt, int lane) {
    const unsigned short* p = wf + ((size_t)((ct * 8 + kt) * 32 + lane)) * 16;
    union { v16bf v; uint4 q[2]; } u;
    u.q[0] = ((const uint4*)p)[0];
    u.q[1] = ((const uint4*)p)[1];
    return u.v;
}

// One 16x16 output tile with A fragments already in registers.
__device__ __forceinline__ v8f gemm_tile_heldA(const v16bf* a,
                                               const unsigned short* wf,
                                               int ct, int lane) {
    v8f acc = {0.f, 0.f, 0.f, 0.f, 0.f, 0.f, 0.f, 0.f};
#pragma unroll
    for (int kt = 0; kt < 8; ++kt) {
        v16bf b = load_b_frag(wf, ct, kt, lane);
        acc = __builtin_amdgcn_wmma_f32_16x16x32_bf16(false, a[kt], false, b,
                                                      (short)0, acc, false, false);
    }
    return acc;
}

// ---------------------------------------------------------------------------
// Kernel 1: repack f32 weight [Cin=256][Cout] -> bf16 B-fragment layout.
// ---------------------------------------------------------------------------
__global__ void repack_kernel(const float* __restrict__ W, int Cout,
                              int ncoltile, unsigned short* __restrict__ dst) {
    int t = blockIdx.x * 256 + threadIdx.x;
    int total = ncoltile * 8 * 512;
    if (t >= total) return;
    int f  = t >> 9;
    int r  = t & 511;
    int l  = r >> 4;
    int e  = r & 15;
    int jt = f >> 3, kt = f & 7;
    int n  = l & 15;
    int k  = kt * 32 + ((e >> 3) << 4) + ((l >> 4) << 3) + (e & 7);
    int j  = jt * 16 + n;
    float v = (j < Cout) ? W[(size_t)k * Cout + j] : 0.0f;
    dst[t] = f32_to_bf16(v);
}

// ---------------------------------------------------------------------------
// Kernel 2: fused encoder. One block = 64 tokens of one batch.
// ---------------------------------------------------------------------------
__global__ __launch_bounds__(256) void fused_enc_kernel(
    const float* __restrict__ s3, const float* __restrict__ s4,
    const float* __restrict__ s5,
    const float* __restrict__ b_score, const float* __restrict__ b1,
    const float* __restrict__ b2, const float* __restrict__ b3,
    const unsigned short* __restrict__ w1f, const unsigned short* __restrict__ w2f,
    const unsigned short* __restrict__ wscoref, const unsigned short* __restrict__ w3f,
    float* __restrict__ enc_logits, float* __restrict__ sig_boxes,
    float* __restrict__ scores_ws) {

    const int nt   = N_TOK / T_TILE;                 // 336 tiles per batch
    const int b    = blockIdx.x / nt;
    const int tok0 = (blockIdx.x % nt) * T_TILE;

    const float* src; int HW; int sp0;
    if (tok0 < NLVL3)               { src = s3 + (size_t)b * C_DIM * NLVL3; HW = NLVL3; sp0 = tok0; }
    else if (tok0 < NLVL3 + NLVL4)  { src = s4 + (size_t)b * C_DIM * NLVL4; HW = NLVL4; sp0 = tok0 - NLVL3; }
    else                            { src = s5 + (size_t)b * C_DIM * NLVL5; HW = NLVL5; sp0 = tok0 - NLVL3 - NLVL4; }

    __shared__ __align__(16) unsigned short lds_a[T_TILE * ROWSTRIDE];
    __shared__ __align__(16) unsigned short lds_h[T_TILE * ROWSTRIDE];
    __shared__ unsigned lds_smax[T_TILE];

    const int tid   = threadIdx.x;
    const int wave  = tid >> 5;
    const int lane  = tid & 31;
    const int lmod  = lane & 15;
    const int lhalf = lane >> 4;

    // ---- stage transposed memory tile: [tok][c] bf16 (vectorized reads) ----
    {
        int i4 = (tid & 15) * 4;                     // 4 consecutive tokens
        int cb = tid >> 4;                           // 16 channels per pass
        for (int c = cb; c < C_DIM; c += 16) {
            const float4 v = *(const float4*)(src + (size_t)c * HW + sp0 + i4);
            lds_a[(i4 + 0) * ROWSTRIDE + c] = f32_to_bf16(v.x);
            lds_a[(i4 + 1) * ROWSTRIDE + c] = f32_to_bf16(v.y);
            lds_a[(i4 + 2) * ROWSTRIDE + c] = f32_to_bf16(v.z);
            lds_a[(i4 + 3) * ROWSTRIDE + c] = f32_to_bf16(v.w);
        }
        if (tid < T_TILE) lds_smax[tid] = 0u;
    }
    __syncthreads();

    // ---- score head: logits = mem @ w_score + b; per-token max key ----
    {
        int rt  = wave & 3;
        v16bf a[8];
        load_a_all(lds_a, rt, lane, a);
        int ct0 = (wave >> 2) ? 3 : 0;
        int ct1 = (wave >> 2) ? 5 : 3;
        size_t tokbase = (size_t)b * N_TOK + tok0 + rt * 16 + lhalf * 8;
#pragma unroll 1
        for (int ct = ct0; ct < ct1; ++ct) {
            v8f acc = gemm_tile_heldA(a, wscoref, ct, lane);
            int   j    = ct * 16 + lmod;
            float bias = b_score[j];
#pragma unroll
            for (int r = 0; r < 8; ++r) {
                float logit = acc[r] + bias;
                enc_logits[(tokbase + r) * NCLS + j] = logit;
                atomicMax(&lds_smax[rt * 16 + lhalf * 8 + r], fkey(logit));
            }
        }
    }
    __syncthreads();
    if (tid < T_TILE)
        scores_ws[(size_t)b * N_TOK + tok0 + tid] = sigmoidf(unkey(lds_smax[tid]));

    // ---- h1 = relu(mem @ w1 + b1) -> lds_h ----
    {
        int rt  = wave >> 1;
        v16bf a[8];
        load_a_all(lds_a, rt, lane, a);
        int ct0 = (wave & 1) * 8;
        int row = (rt * 16 + lhalf * 8) * ROWSTRIDE;
#pragma unroll 1
        for (int ct = ct0; ct < ct0 + 8; ++ct) {
            v8f acc = gemm_tile_heldA(a, w1f, ct, lane);
            int   j    = ct * 16 + lmod;
            float bias = b1[j];
#pragma unroll
            for (int r = 0; r < 8; ++r) {
                float h = acc[r] + bias;
                h = h > 0.f ? h : 0.f;
                lds_h[row + r * ROWSTRIDE + j] = f32_to_bf16(h);
            }
        }
    }
    __syncthreads();

    // ---- h2 = relu(h1 @ w2 + b2) -> lds_a (reuse) ----
    {
        int rt  = wave >> 1;
        v16bf a[8];
        load_a_all(lds_h, rt, lane, a);
        int ct0 = (wave & 1) * 8;
        int row = (rt * 16 + lhalf * 8) * ROWSTRIDE;
#pragma unroll 1
        for (int ct = ct0; ct < ct0 + 8; ++ct) {
            v8f acc = gemm_tile_heldA(a, w2f, ct, lane);
            int   j    = ct * 16 + lmod;
            float bias = b2[j];
#pragma unroll
            for (int r = 0; r < 8; ++r) {
                float h = acc[r] + bias;
                h = h > 0.f ? h : 0.f;
                lds_a[row + r * ROWSTRIDE + j] = f32_to_bf16(h);
            }
        }
    }
    __syncthreads();

    // ---- boxes = sigmoid(h2 @ w3 + b3) ----
    if (wave < 4) {
        int rt = wave;
        v16bf a[8];
        load_a_all(lds_a, rt, lane, a);
        v8f acc = gemm_tile_heldA(a, w3f, 0, lane);
        if (lmod < 4) {
            float bias = b3[lmod];
            size_t tokbase = (size_t)b * N_TOK + tok0 + rt * 16 + lhalf * 8;
#pragma unroll
            for (int r = 0; r < 8; ++r)
                sig_boxes[(tokbase + r) * 4 + lmod] = sigmoidf(acc[r] + bias);
        }
    }
}

// ---------------------------------------------------------------------------
// Kernel 3: per-batch top-300 (radix select + bitonic sort).
// ---------------------------------------------------------------------------
__global__ __launch_bounds__(1024) void topk_kernel(
    const float* __restrict__ scores, int* __restrict__ idx_ws,
    float* __restrict__ out_idx, float* __restrict__ out_scr) {

    const int b   = blockIdx.x;
    const int tid = threadIdx.x;
    const float* s = scores + (size_t)b * N_TOK;

    __shared__ unsigned hist[256];
    __shared__ unsigned long long pairs[512];
    __shared__ unsigned sh_prefix, sh_need, sh_cntG, sh_cntE;

    if (tid == 0) { sh_prefix = 0u; sh_need = KQ; }
    __syncthreads();

    for (int pass = 3; pass >= 0; --pass) {
        if (tid < 256) hist[tid] = 0u;
        __syncthreads();
        unsigned prefix = sh_prefix;
        unsigned hm = (pass == 3) ? 0u : (0xFFFFFFFFu << ((pass + 1) * 8));
        for (int n = tid; n < N_TOK; n += 1024) {
            unsigned key = fkey(s[n]);
            if ((key & hm) == prefix)
                atomicAdd(&hist[(key >> (pass * 8)) & 255u], 1u);
        }
        __syncthreads();
        if (tid == 0) {
            unsigned need = sh_need, cum = 0u; int chosen = 0;
            for (int v = 255; v >= 0; --v) {
                unsigned c = hist[v];
                if (cum + c >= need) { chosen = v; sh_need = need - cum; break; }
                cum += c;
            }
            sh_prefix = prefix | ((unsigned)chosen << (pass * 8));
        }
        __syncthreads();
    }
    const unsigned T = sh_prefix;
    if (tid == 0) { sh_cntG = 0u; sh_cntE = 0u; }
    __syncthreads();

    for (int n = tid; n < N_TOK; n += 1024) {
        unsigned key = fkey(s[n]);
        if (key > T) {
            unsigned p = atomicAdd(&sh_cntG, 1u);
            pairs[p] = ((unsigned long long)key << 32) | (unsigned)(0xFFFFFFFFu - (unsigned)n);
        }
    }
    __syncthreads();
    const unsigned G = sh_cntG;
    for (int n = tid; n < N_TOK; n += 1024) {
        unsigned key = fkey(s[n]);
        if (key == T) {
            unsigned p = G + atomicAdd(&sh_cntE, 1u);
            if (p < KQ)
                pairs[p] = ((unsigned long long)key << 32) | (unsigned)(0xFFFFFFFFu - (unsigned)n);
        }
    }
    __syncthreads();
    if (tid < 512 && tid >= KQ) pairs[tid] = 0ull;

    // bitonic sort, descending (tie -> smaller index first via inverted low word)
    for (unsigned k = 2; k <= 512; k <<= 1) {
        for (unsigned jj = k >> 1; jj > 0; jj >>= 1) {
            __syncthreads();
            if (tid < 512) {
                unsigned i = tid, ixj = i ^ jj;
                if (ixj > i) {
                    unsigned long long a = pairs[i], c = pairs[ixj];
                    bool up = ((i & k) == 0);
                    if (up ? (a < c) : (a > c)) { pairs[i] = c; pairs[ixj] = a; }
                }
            }
        }
    }
    __syncthreads();

    if (tid < KQ) {
        unsigned long long pr = pairs[tid];
        unsigned key = (unsigned)(pr >> 32);
        unsigned n   = 0xFFFFFFFFu - (unsigned)(pr & 0xFFFFFFFFull);
        idx_ws[b * KQ + tid]  = (int)n;
        out_idx[b * KQ + tid] = (float)n;
        out_scr[b * KQ + tid] = unkey(key);
    }
}

// ---------------------------------------------------------------------------
// Kernel 4: tgt = gather(memory) @ w_proj + b_proj ; ref_points gather.
// ---------------------------------------------------------------------------
__global__ __launch_bounds__(256) void tgt_kernel(
    const float* __restrict__ s3, const float* __restrict__ s4,
    const float* __restrict__ s5, const float* __restrict__ b_proj,
    const unsigned short* __restrict__ wprojf, const int* __restrict__ idx_ws,
    const float* __restrict__ sig_boxes, float* __restrict__ tgt,
    float* __restrict__ refp) {

    const int b = blockIdx.x;
    __shared__ __align__(16) unsigned short lds_a[T_TILE * ROWSTRIDE];
    __shared__ int qidx[T_TILE];

    const int tid   = threadIdx.x;
    const int wave  = tid >> 5;
    const int lane  = tid & 31;
    const int lmod  = lane & 15;
    const int lhalf = lane >> 4;

    for (int chunk = 0; chunk < 5; ++chunk) {            // 5*64 = 320 >= 300
        __syncthreads();
        if (tid < T_TILE) {
            int q = chunk * T_TILE + tid;
            qidx[tid] = (q < KQ) ? idx_ws[b * KQ + q] : 0;
        }
        __syncthreads();
        {
            int i  = tid & 63;
            int cb = tid >> 6;
            int n  = qidx[i];
            const float* src; int HW; int sp;
            if (n < NLVL3)              { src = s3 + (size_t)b * C_DIM * NLVL3; HW = NLVL3; sp = n; }
            else if (n < NLVL3 + NLVL4) { src = s4 + (size_t)b * C_DIM * NLVL4; HW = NLVL4; sp = n - NLVL3; }
            else                        { src = s5 + (size_t)b * C_DIM * NLVL5; HW = NLVL5; sp = n - NLVL3 - NLVL4; }
            for (int c = cb; c < C_DIM; c += 4)
                lds_a[i * ROWSTRIDE + c] = f32_to_bf16(src[(size_t)c * HW + sp]);
        }
        __syncthreads();
        {
            int rt  = wave >> 1;
            v16bf a[8];
            load_a_all(lds_a, rt, lane, a);
            int ct0 = (wave & 1) * 8;
            int qbase = chunk * T_TILE + rt * 16 + lhalf * 8;
#pragma unroll 1
            for (int ct = ct0; ct < ct0 + 8; ++ct) {
                v8f acc = gemm_tile_heldA(a, wprojf, ct, lane);
                int   j    = ct * 16 + lmod;
                float bias = b_proj[j];
#pragma unroll
                for (int r = 0; r < 8; ++r) {
                    int q = qbase + r;
                    if (q < KQ) tgt[((size_t)b * KQ + q) * C_DIM + j] = acc[r] + bias;
                }
            }
        }
    }
    __syncthreads();
    for (int q = tid; q < KQ; q += 256) {
        int n = idx_ws[b * KQ + q];
        size_t sb = ((size_t)b * N_TOK + n) * 4;
        size_t db = ((size_t)b * KQ + q) * 4;
        refp[db + 0] = sig_boxes[sb + 0];
        refp[db + 1] = sig_boxes[sb + 1];
        refp[db + 2] = sig_boxes[sb + 2];
        refp[db + 3] = sig_boxes[sb + 3];
    }
}

// ---------------------------------------------------------------------------
extern "C" void kernel_launch(void* const* d_in, const int* in_sizes, int n_in,
                              void* d_out, int out_size, void* d_ws, size_t ws_size,
                              hipStream_t stream) {
    const float* s3      = (const float*)d_in[0];
    const float* s4      = (const float*)d_in[1];
    const float* s5      = (const float*)d_in[2];
    const float* w_score = (const float*)d_in[3];
    const float* b_score = (const float*)d_in[4];
    const float* w1      = (const float*)d_in[5];
    const float* b1      = (const float*)d_in[6];
    const float* w2      = (const float*)d_in[7];
    const float* b2      = (const float*)d_in[8];
    const float* w3      = (const float*)d_in[9];
    const float* b3      = (const float*)d_in[10];
    const float* w_proj  = (const float*)d_in[11];
    const float* b_proj  = (const float*)d_in[12];

    char* ws = (char*)d_ws;
    unsigned short* w1f     = (unsigned short*)(ws + WS_W1);
    unsigned short* w2f     = (unsigned short*)(ws + WS_W2);
    unsigned short* wscoref = (unsigned short*)(ws + WS_WSC);
    unsigned short* w3f     = (unsigned short*)(ws + WS_W3);
    unsigned short* wprojf  = (unsigned short*)(ws + WS_WPROJ);
    float*          scoresW = (float*)(ws + WS_SCORES);
    int*            idxW    = (int*)(ws + WS_IDX);

    float* out = (float*)d_out;
    float* out_tgt = out + OFF_TGT;
    float* out_ref = out + OFF_REF;
    float* out_log = out + OFF_LOG;
    float* out_sbx = out + OFF_SBX;
    float* out_idx = out + OFF_IDX;
    float* out_scr = out + OFF_SCR;

    // 1) weight repack (B-fragment bf16 layout)
    repack_kernel<<<256, 256, 0, stream>>>(w1, 256, 16, w1f);
    repack_kernel<<<256, 256, 0, stream>>>(w2, 256, 16, w2f);
    repack_kernel<<<80,  256, 0, stream>>>(w_score, 80, 5, wscoref);
    repack_kernel<<<16,  256, 0, stream>>>(w3, 4, 1, w3f);
    repack_kernel<<<256, 256, 0, stream>>>(w_proj, 256, 16, wprojf);

    // 2) fused encoder over all tokens
    fused_enc_kernel<<<BATCH * (N_TOK / T_TILE), 256, 0, stream>>>(
        s3, s4, s5, b_score, b1, b2, b3, w1f, w2f, wscoref, w3f,
        out_log, out_sbx, scoresW);

    // 3) per-batch top-300
    topk_kernel<<<BATCH, 1024, 0, stream>>>(scoresW, idxW, out_idx, out_scr);

    // 4) query projection + ref point gather
    tgt_kernel<<<BATCH, 256, 0, stream>>>(s3, s4, s5, b_proj, wprojf, idxW,
                                          out_sbx, out_tgt, out_ref);
}
